// GaugeTheoryNode_35253091566215
// MI455X (gfx1250) — compile-verified, matrix operand order checked by reading, not run on hardware
//
#include <hip/hip_runtime.h>
#include <hip/hip_bf16.h>
#include <stdint.h>

// ---------------------------------------------------------------------------
// GaugeTheoryNode pipeline for MI455X (gfx1250, wave32, WMMA, async-LDS DMA).
// HBM-bound workload (~900 GFLOP vs ~650MB traffic): bf16 WMMA GEMMs with
// f32 accumulation, double-buffered LDS fed by GLOBAL_LOAD_ASYNC_TO_LDS_B128
// when the toolchain exposes it (ASYNCcnt-tracked), sync fallback otherwise.
// ---------------------------------------------------------------------------

typedef __attribute__((ext_vector_type(16))) __bf16 v16bf;
typedef __attribute__((ext_vector_type(8)))  float  v8f;

union Frag {
    v16bf v;
    uint4 q[2];
};

#if defined(__AMDGCN__) && __has_builtin(__builtin_amdgcn_global_load_async_to_lds_b128)
#define HAVE_ASYNC_LDS 1
#else
#define HAVE_ASYNC_LDS 0
#endif

// gcc-style vector matching the builtin's printed parameter type:
//   "__attribute__((__vector_size__(4 * sizeof(int)))) int __device__ *"
typedef int v4i_vec __attribute__((vector_size(16)));
#define AS_GLOBAL __attribute__((address_space(1)))
#define AS_LOCAL  __attribute__((address_space(3)))

__device__ __forceinline__ void async_copy_b128(const void* g, void* l) {
#if HAVE_ASYNC_LDS
    // global side: flat and AS1 addresses coincide numerically -> go via int
    AS_GLOBAL v4i_vec* gp = (AS_GLOBAL v4i_vec*)(uintptr_t)g;
    // LDS side: true addrspacecast generic -> local
    AS_LOCAL v4i_vec* lp = (AS_LOCAL v4i_vec*)l;
    __builtin_amdgcn_global_load_async_to_lds_b128(gp, lp, 0, 0);
#else
    *(uint4*)l = *(const uint4*)g;   // sync fallback
#endif
}

__device__ __forceinline__ void wait_async_lds() {
#if HAVE_ASYNC_LDS
#if __has_builtin(__builtin_amdgcn_s_wait_asynccnt)
    __builtin_amdgcn_s_wait_asynccnt(0);
#else
    asm volatile("s_wait_asynccnt 0" ::: "memory");
#endif
#endif
}

// round-to-nearest-even f32 -> bf16 (bit pattern as u16)
__device__ __forceinline__ unsigned short f2bf(float f) {
    union { float f; unsigned int u; } cv; cv.f = f;
    unsigned int u = cv.u;
    u += 0x7FFFu + ((u >> 16) & 1u);
    return (unsigned short)(u >> 16);
}

__device__ __forceinline__ float bf2f(unsigned short s) {
    union { unsigned int u; float f; } cv;
    cv.u = ((unsigned int)s) << 16;
    return cv.f;
}

// ---------------------------------------------------------------------------
// Tiled weight transpose + f32->bf16:  W (K x N, f32) -> Wt (N x K, bf16).
// 32x32 tiles through LDS; coalesced reads and writes.
// ---------------------------------------------------------------------------
__global__ __launch_bounds__(256) void transpose_conv_kernel(
    const float* __restrict__ W, unsigned short* __restrict__ Wt, int K, int N) {
    __shared__ unsigned short tile[32][33];
    const int tx = threadIdx.x & 31;
    const int ty = threadIdx.x >> 5;          // 0..7
    const int k0 = blockIdx.y * 32;
    const int n0 = blockIdx.x * 32;
#pragma unroll
    for (int i = 0; i < 4; ++i) {
        const int k = k0 + ty + i * 8;
        tile[ty + i * 8][tx] = f2bf(W[(size_t)k * N + n0 + tx]);
    }
    __syncthreads();
#pragma unroll
    for (int i = 0; i < 4; ++i) {
        const int n = n0 + ty + i * 8;
        Wt[(size_t)n * K + k0 + tx] = tile[tx][ty + i * 8];
    }
}

// ---------------------------------------------------------------------------
// fc1: h = relu(x[:, :4] @ fw1 + fb1), stored as bf16.  K=4 -> VALU kernel.
// ---------------------------------------------------------------------------
__global__ __launch_bounds__(256) void fc1_kernel(
    const float* __restrict__ x, const float* __restrict__ w,
    const float* __restrict__ b, unsigned short* __restrict__ h) {
    int idx = blockIdx.x * 256 + threadIdx.x;     // over 16384*2048
    int col = idx & 2047;
    int row = idx >> 11;
    float acc = b[col];
#pragma unroll
    for (int k = 0; k < 4; ++k) acc += x[row * 4 + k] * w[k * 2048 + col];
    acc = fmaxf(acc, 0.0f);
    h[idx] = f2bf(acc);
}

// ---------------------------------------------------------------------------
// Tiled WMMA bf16 GEMM:  C = act(A @ Bt^T + bias)
//   A:  M x K  (f32 when A_F32, else bf16; row-major)
//   Bt: N x K  (bf16, pre-transposed -> K-contiguous fragment reads)
// BM=BN=128, BK=32; 256 threads = 8 waves in 2x4; each wave owns 64x32 of C
// (4x2 tiles) -> 8 v_wmma_f32_16x16x32_bf16 per wave per K-step.
// Double-buffered LDS; bf16 tiles staged with async global->LDS DMA.
// ---------------------------------------------------------------------------
#define BM 128
#define BN 128
#define BK 32
#define LDT 40   // padded LDS row pitch in ushorts (80B) to spread banks

template <int A_F32>
__global__ __launch_bounds__(256) void gemm_bf16_wmma(
    const void* __restrict__ Aptr,
    const unsigned short* __restrict__ Bt,
    const float* __restrict__ bias,
    float* __restrict__ Cf,            // optional f32 output (nullptr to skip)
    unsigned short* __restrict__ Cb,   // optional bf16 output (nullptr to skip)
    int M, int N, int K, int relu) {
    __shared__ unsigned short As[2][BM * LDT];
    __shared__ unsigned short Bs[2][BN * LDT];

    const int tid  = threadIdx.x;
    const int wid  = tid >> 5;
    const int lane = tid & 31;
    const int bm = blockIdx.y * BM;
    const int bn = blockIdx.x * BN;
    const int wm = (wid >> 2) * 64;
    const int wn = (wid & 3) * 32;
    const int lm = lane & 15;
    const int lh = lane >> 4;

    const float*          Af = (const float*)Aptr;
    const unsigned short* Ah = (const unsigned short*)Aptr;

    // ---- staging helpers ---------------------------------------------------
    auto stageA = [&](int buf, int kk) {
        if (A_F32) {
            // convert f32 -> bf16 in flight (sync path; DMA can't convert)
            const int r0 = tid >> 3;
            const int c0 = (tid & 7) * 4;
#pragma unroll
            for (int p = 0; p < 4; ++p) {
                const int row = r0 + p * 32;
                const float* src = Af + (size_t)(bm + row) * K + kk + c0;
                __builtin_prefetch(src + BK, 0, 1);   // speculative: safe
                const float4 fv = *(const float4*)src;
                uint2 pk;
                pk.x = (unsigned int)f2bf(fv.x) | ((unsigned int)f2bf(fv.y) << 16);
                pk.y = (unsigned int)f2bf(fv.z) | ((unsigned int)f2bf(fv.w) << 16);
                *(uint2*)&As[buf][row * LDT + c0] = pk;
            }
        } else {
            const int r0 = tid >> 2;
            const int c0 = (tid & 3) * 8;
#pragma unroll
            for (int p = 0; p < 2; ++p) {
                const int row = r0 + p * 64;
                async_copy_b128(Ah + (size_t)(bm + row) * K + kk + c0,
                                &As[buf][row * LDT + c0]);
            }
        }
    };
    auto stageB = [&](int buf, int kk) {
        const int r0 = tid >> 2;
        const int c0 = (tid & 3) * 8;
#pragma unroll
        for (int p = 0; p < 2; ++p) {
            const int row = r0 + p * 64;
            async_copy_b128(Bt + (size_t)(bn + row) * K + kk + c0,
                            &Bs[buf][row * LDT + c0]);
        }
    };

    v8f acc[4][2];
#pragma unroll
    for (int i = 0; i < 4; ++i)
#pragma unroll
        for (int j = 0; j < 2; ++j)
#pragma unroll
            for (int r = 0; r < 8; ++r) acc[i][j][r] = 0.0f;

    // prologue: fill buffer 0
    stageA(0, 0);
    stageB(0, 0);

    int cur = 0;
    for (int kk = 0; kk < K; kk += BK) {
        wait_async_lds();       // my wave's DMA into buf[cur] done
        __syncthreads();        // everyone's staging of buf[cur] visible

        if (kk + BK < K) {      // overlap: DMA next tile into other buffer
            stageA(cur ^ 1, kk + BK);
            stageB(cur ^ 1, kk + BK);
        }

        // ---- fragments per the 16-bit A/B VGPR layout:
        // lane half lh: K-runs [lh*8, lh*8+8) and [16+lh*8, 16+lh*8+8) ----
        Frag a[4], b[2];
#pragma unroll
        for (int i = 0; i < 4; ++i) {
            const int row = wm + i * 16 + lm;
            a[i].q[0] = *(const uint4*)&As[cur][row * LDT + lh * 8];
            a[i].q[1] = *(const uint4*)&As[cur][row * LDT + 16 + lh * 8];
        }
#pragma unroll
        for (int j = 0; j < 2; ++j) {
            const int row = wn + j * 16 + lm;
            b[j].q[0] = *(const uint4*)&Bs[cur][row * LDT + lh * 8];
            b[j].q[1] = *(const uint4*)&Bs[cur][row * LDT + 16 + lh * 8];
        }
#pragma unroll
        for (int i = 0; i < 4; ++i)
#pragma unroll
            for (int j = 0; j < 2; ++j)
                acc[i][j] = __builtin_amdgcn_wmma_f32_16x16x32_bf16(
                    false, a[i].v, false, b[j].v, (short)0, acc[i][j],
                    false, false);
        cur ^= 1;
    }

    // ---- epilogue: C/D layout -> row = i*16 + r + lh*8, col = lm ----
#pragma unroll
    for (int i = 0; i < 4; ++i) {
#pragma unroll
        for (int j = 0; j < 2; ++j) {
            const int colg = bn + wn + j * 16 + lm;
            const float bv = bias ? bias[colg] : 0.0f;
#pragma unroll
            for (int r = 0; r < 8; ++r) {
                const int rowg = bm + wm + i * 16 + r + lh * 8;
                float v = acc[i][j][r] + bv;
                if (relu) v = fmaxf(v, 0.0f);
                const size_t o = (size_t)rowg * N + colg;
                if (Cf) Cf[o] = v;
                if (Cb) Cb[o] = f2bf(v);
            }
        }
    }
}

// ---------------------------------------------------------------------------
// GEMV head: out[row] = sum_k bf16(A[row,k]) * w[k] + b[0]; one wave per row.
// ---------------------------------------------------------------------------
__global__ __launch_bounds__(256) void gemv_kernel(
    const unsigned short* __restrict__ A, const float* __restrict__ w,
    const float* __restrict__ b, float* __restrict__ out, int M, int K) {
    const int wid  = threadIdx.x >> 5;
    const int lane = threadIdx.x & 31;
    const int row  = blockIdx.x * 8 + wid;
    if (row >= M) return;
    float acc = 0.0f;
    for (int k = lane; k < K; k += 32)
        acc += bf2f(A[(size_t)row * K + k]) * w[k];
#pragma unroll
    for (int off = 16; off > 0; off >>= 1)
        acc += __shfl_xor(acc, off, 32);
    if (lane == 0) out[row] = acc + b[0];
}

// ---------------------------------------------------------------------------
// Wilson plaquette (exact reference semantics, single thread).
// ---------------------------------------------------------------------------
__global__ void plaq_kernel(const float* __restrict__ x,
                            const float* __restrict__ field,
                            float* __restrict__ out) {
    if (threadIdx.x != 0 || blockIdx.x != 0) return;
    int c[4];
#pragma unroll
    for (int i = 0; i < 4; ++i) {
        int v = (int)x[i];           // trunc toward zero (astype int32)
        v %= 10; if (v < 0) v += 10; // python-style mod L
        c[i] = v;
    }
    float loop = 1.0f;
    for (int mu = 0; mu < 4; ++mu) {
        for (int nu = mu + 1; nu < 4; ++nu) {
            int p[4] = {c[0], c[1], c[2], c[3]};
            float l1 = field[((p[0] * 10 + p[1]) * 10 + p[2]) * 4 + mu];
            p[mu] = (p[mu] + 1) % 10;
            float l2 = field[((p[0] * 10 + p[1]) * 10 + p[2]) * 4 + nu];
            p[nu] = (p[nu] + 1) % 10;
            float l3 = -field[((p[0] * 10 + p[1]) * 10 + p[2]) * 4 + mu];
            float l4 = -field[((c[0] * 10 + c[1]) * 10 + c[2]) * 4 + nu];
            loop *= expf(l1 + l2 + l3 + l4);
        }
    }
    *out = loop;
}

// ---------------------------------------------------------------------------
// Host-side orchestration.
// ---------------------------------------------------------------------------
extern "C" void kernel_launch(void* const* d_in, const int* in_sizes, int n_in,
                              void* d_out, int out_size, void* d_ws, size_t ws_size,
                              hipStream_t stream) {
    const float* x    = (const float*)d_in[0];
    const float* fld  = (const float*)d_in[1];
    const float* fw1  = (const float*)d_in[2];
    const float* fb1  = (const float*)d_in[3];
    const float* fw2  = (const float*)d_in[4];
    const float* fb2  = (const float*)d_in[5];
    const float* ww1  = (const float*)d_in[6];
    const float* wb1  = (const float*)d_in[7];
    const float* ww2  = (const float*)d_in[8];
    const float* wb2  = (const float*)d_in[9];
    const float* ww3  = (const float*)d_in[10];
    const float* wb3  = (const float*)d_in[11];
    const float* tw1  = (const float*)d_in[12];
    const float* tb1  = (const float*)d_in[13];
    const float* tw2  = (const float*)d_in[14];
    const float* tb2  = (const float*)d_in[15];
    float* out = (float*)d_out;

    const int M = 16384;
    const size_t FO = (size_t)M * 4096;   // field_output elements

    // workspace carve-up (bf16 buffers, 16B-aligned by construction)
    char* ws = (char*)d_ws;
    unsigned short* fw2T = (unsigned short*)ws; ws += (size_t)4096 * 2048 * 2; // 16MB
    unsigned short* ww1T = (unsigned short*)ws; ws += (size_t)2048 * 4096 * 2; // 16MB
    unsigned short* tw1T = (unsigned short*)ws; ws += (size_t)2048 * 4096 * 2; // 16MB
    unsigned short* ww2T = (unsigned short*)ws; ws += (size_t)1024 * 2048 * 2; //  4MB
    unsigned short* hbuf = (unsigned short*)ws; ws += (size_t)M * 2048 * 2;    // 64MB
    unsigned short* wbuf = (unsigned short*)ws; ws += (size_t)M * 2048 * 2;    // 64MB (w1, reused for t)
    unsigned short* w2b  = (unsigned short*)ws; ws += (size_t)M * 1024 * 2;    // 32MB

    // 1) weight transposes (f32 -> bf16, N x K), 32x32 LDS tiles
    transpose_conv_kernel<<<dim3(4096 / 32, 2048 / 32), 256, 0, stream>>>(fw2, fw2T, 2048, 4096);
    transpose_conv_kernel<<<dim3(2048 / 32, 4096 / 32), 256, 0, stream>>>(ww1, ww1T, 4096, 2048);
    transpose_conv_kernel<<<dim3(2048 / 32, 4096 / 32), 256, 0, stream>>>(tw1, tw1T, 4096, 2048);
    transpose_conv_kernel<<<dim3(1024 / 32, 2048 / 32), 256, 0, stream>>>(ww2, ww2T, 2048, 1024);

    // 2) h = relu(x @ fw1 + fb1)  -> bf16
    fc1_kernel<<<((size_t)M * 2048) / 256, 256, 0, stream>>>(x, fw1, fb1, hbuf);

    // 3) field_output = h @ fw2 + fb2  -> f32 in d_out
    gemm_bf16_wmma<0><<<dim3(4096 / BN, M / BM), 256, 0, stream>>>(
        hbuf, fw2T, fb2, out, nullptr, M, 4096, 2048, 0);

    // 4) w1 = relu(field_output @ ww1 + wb1) -> bf16
    gemm_bf16_wmma<1><<<dim3(2048 / BN, M / BM), 256, 0, stream>>>(
        out, ww1T, wb1, nullptr, wbuf, M, 2048, 4096, 1);

    // 5) w2 = relu(w1 @ ww2 + wb2) -> bf16
    gemm_bf16_wmma<0><<<dim3(1024 / BN, M / BM), 256, 0, stream>>>(
        wbuf, ww2T, wb2, nullptr, w2b, M, 1024, 2048, 1);

    // 6) wilson_output = w2 @ ww3 + wb3
    gemv_kernel<<<M / 8, 256, 0, stream>>>(w2b, ww3, wb3, out + FO, M, 1024);

    // 7) t = relu(field_output @ tw1 + tb1) -> bf16 (reuse w1 buffer)
    gemm_bf16_wmma<1><<<dim3(2048 / BN, M / BM), 256, 0, stream>>>(
        out, tw1T, tb1, nullptr, wbuf, M, 2048, 4096, 1);

    // 8) topology_output = t @ tw2 + tb2
    gemv_kernel<<<M / 8, 256, 0, stream>>>(wbuf, tw2, tb2, out + FO + M, M, 2048);

    // 9) plaquette scalar
    plaq_kernel<<<1, 32, 0, stream>>>(x, fld, out + FO + 2 * M);

    (void)in_sizes; (void)n_in; (void)out_size; (void)ws_size;
}
